// ExpertChoiceMoEMatcher_89043261981219
// MI455X (gfx1250) — matrix operand order repaired and checked by reading, not compiled, not run on hardware
//
#include <hip/hip_runtime.h>
#include <stdint.h>

#define B_T   16384
#define DDIM  512
#define NEXP  16
#define KSEL  1024

// d_out layout (floats)
#define RES_OFF 0
#define IDX_OFF 16777216      // B_T*DDIM*2
#define SC_OFF  16793600      // + NEXP*KSEL
#define CNT_OFF 16809984      // + NEXP*KSEL

typedef __attribute__((ext_vector_type(16))) __bf16 v16bf;
typedef __attribute__((ext_vector_type(8)))  float  v8f;
typedef __attribute__((ext_vector_type(8)))  int    v8i;

__device__ __forceinline__ unsigned short f2bf(float f) {
    union { float f; unsigned int u; } v; v.f = f;
    unsigned int u = v.u;
    unsigned int r = u + 0x7FFFu + ((u >> 16) & 1u);   // round-to-nearest-even
    return (unsigned short)(r >> 16);
}

// ---------------- 1) deinterleave x -> bf16 planes Xr, Xi -------------------
__global__ void cvt_x_kernel(const float* __restrict__ x,
                             unsigned short* __restrict__ Xr,
                             unsigned short* __restrict__ Xi) {
    int i = blockIdx.x * 256 + threadIdx.x;          // i < B_T*DDIM
    float2 v = ((const float2*)x)[i];
    Xr[i] = f2bf(v.x);
    Xi[i] = f2bf(v.y);
}

// ------- 2) deinterleave + transpose experts_weight -> WrT/WiT [e][f][d] ----
__global__ void cvt_w_kernel(const float* __restrict__ w,
                             unsigned short* __restrict__ WrT,
                             unsigned short* __restrict__ WiT) {
    int i = blockIdx.x * 256 + threadIdx.x;          // i < E*D*D, pair idx (e,d,f)
    int e = i >> 18;                                  // / (512*512)
    int r = i & 262143;
    int d = r >> 9;
    int f = r & 511;
    float2 v = ((const float2*)w)[i];
    int o = e * 262144 + f * 512 + d;                 // transposed: contiguous in d(=K)
    WrT[o] = f2bf(v.x);
    WiT[o] = f2bf(v.y);
}

// ---------------- 3) gating: scores[t][e] = x_flat[t] . gw[:,e] -------------
__global__ void gate_kernel(const float* __restrict__ x,
                            const float* __restrict__ gw,
                            float* __restrict__ scores) {
    int tok = blockIdx.x * 256 + threadIdx.x;        // one token per thread
    const float* xr = x + (size_t)tok * 1024;
    float acc[NEXP];
#pragma unroll
    for (int e = 0; e < NEXP; ++e) acc[e] = 0.0f;
    for (int k = 0; k < 1024; ++k) {
        float xv = xr[k];
        const float* g = gw + k * NEXP;
#pragma unroll
        for (int e = 0; e < NEXP; ++e) acc[e] = fmaf(xv, g[e], acc[e]);
    }
#pragma unroll
    for (int e = 0; e < NEXP; ++e) scores[tok * NEXP + e] = acc[e];
}

// ------- 4) per-expert full bitonic sort (descending) in 128KB LDS ----------
__global__ void topk_kernel(const float* __restrict__ scores,
                            float* __restrict__ out,
                            float* __restrict__ counts) {
    extern __shared__ unsigned char smem[];
    float* s  = (float*)smem;
    int*   id = (int*)(smem + B_T * sizeof(float));
    const int e   = blockIdx.x;
    const int tid = threadIdx.x;

    for (int i = tid; i < B_T; i += 1024) { s[i] = scores[i * NEXP + e]; id[i] = i; }
    __syncthreads();

    for (int size = 2; size <= B_T; size <<= 1) {
        for (int stride = size >> 1; stride > 0; stride >>= 1) {
            for (int i = tid; i < B_T; i += 1024) {
                int p = i ^ stride;
                if (p > i) {
                    float a = s[i], b = s[p];
                    bool descBlock = (i & size) == 0;
                    bool sw = descBlock ? (a < b) : (a > b);
                    if (sw) {
                        s[i] = b; s[p] = a;
                        int t = id[i]; id[i] = id[p]; id[p] = t;
                    }
                }
            }
            __syncthreads();
        }
    }
    if (tid < KSEL) {
        int ix = id[tid];
        ((int*)out)[IDX_OFF + e * KSEL + tid] = ix;
        out[SC_OFF + e * KSEL + tid] = s[tid];
        atomicAdd(&counts[ix], 1.0f);
    }
}

// ---- 5) per-expert complex GEMM: block = 16 rows x 512 cols ----------------
//  A (gathered, both planes, 32KB bf16) staged in LDS via async-to-LDS copy.
//  Each of 8 waves register-blocks 4 N-tiles (16x64), reusing the A fragment.
//  The 4 complex sub-products run as 4 independent WMMA accumulation chains
//  (no VALU in the K-loop -> no WMMA/VALU WAR hazard bubbles); combined once
//  in the epilogue: accR = P1 - P2, accI = P3 + P4.
__global__ void __launch_bounds__(256)
moe_gemm_kernel(const unsigned short* __restrict__ Xr,
                const unsigned short* __restrict__ Xi,
                const unsigned short* __restrict__ WrT,
                const unsigned short* __restrict__ WiT,
                float* __restrict__ out) {
    __shared__ __align__(16) unsigned short sA[2 * 16 * 512];   // Ar | Ai, 32 KB

    const int tid    = threadIdx.x;
    const int lane   = tid & 31;
    const int waveId = tid >> 5;
    const int e  = blockIdx.x >> 6;                    // expert
    const int mt = blockIdx.x & 63;                    // 16-row tile within expert

    const int*   idx_all = ((const int*)out) + IDX_OFF;
    const float* sc_all  = out + SC_OFF;

    // ---- async-copy 16 gathered token rows (both planes) into LDS ----------
    const unsigned sA_base = (unsigned)(uintptr_t)sA;  // low 32 bits = LDS offset
#pragma unroll
    for (int it = 0; it < 8; ++it) {
        int c     = it * 256 + tid;                    // 2048 chunks of 16B
        int plane = c >> 10;                           // uniform per iteration
        int row   = (c >> 6) & 15;
        int off   = c & 63;                            // 16B units in row
        int token = idx_all[e * KSEL + mt * 16 + row];
        const unsigned short* src =
            (plane ? Xi : Xr) + (size_t)token * 512 + off * 8;
        unsigned dst = sA_base + (unsigned)(plane * 16384 + row * 1024 + off * 16);
        asm volatile("global_load_async_to_lds_b128 %0, %1, off"
                     :: "v"(dst), "v"((unsigned long long)(uintptr_t)src)
                     : "memory");
    }
    asm volatile("s_wait_asynccnt 0" ::: "memory");
    __syncthreads();

    const int row   = lane & 15;
    const int ahalf = (lane >> 4) * 8;
    const int bhalf = (lane >> 4) * 16;
    const int* sA32 = (const int*)sA;                  // dword view of LDS

    // 4 product chains x 4 N-tiles
    v8f p1[4], p2[4], p3[4], p4[4];
    const v8f vzero = {0.f,0.f,0.f,0.f,0.f,0.f,0.f,0.f};
#pragma unroll
    for (int t = 0; t < 4; ++t) { p1[t]=vzero; p2[t]=vzero; p3[t]=vzero; p4[t]=vzero; }

    for (int kstep = 0; kstep < 16; ++kstep) {
        const int kbase = kstep * 32;

        // A fragments from LDS (shared by all 4 N-tiles of this wave)
        v8i ar, ai;
#pragma unroll
        for (int j = 0; j < 8; ++j) {
            const int ka = kbase + ahalf + ((j < 4) ? 2 * j : 2 * j + 8);
            ar[j] = sA32[row * 256 + (ka >> 1)];
            ai[j] = sA32[4096 + row * 256 + (ka >> 1)];
        }
        v16bf fAr = __builtin_bit_cast(v16bf, ar);
        v16bf fAi = __builtin_bit_cast(v16bf, ai);

#pragma unroll
        for (int t = 0; t < 4; ++t) {
            const int ncol = (waveId * 4 + t) * 16 + (lane & 15);
            const unsigned short* wr = WrT + (size_t)e * 262144 + (size_t)ncol * 512;
            const unsigned short* wi = WiT + (size_t)e * 262144 + (size_t)ncol * 512;
            if (t == 0 && kstep + 1 < 16) {            // -> global_prefetch_b8
                __builtin_prefetch((const void*)(wr + kbase + 32), 0, 3);
                __builtin_prefetch((const void*)(wi + kbase + 32), 0, 3);
            }
            v8i br, bi;
#pragma unroll
            for (int j = 0; j < 8; ++j) {
                const int kb = kbase + ahalf + ((j < 4) ? 2 * j : 2 * j + 8);
                br[j] = *(const int*)(wr + kb);
                bi[j] = *(const int*)(wi + kb);
            }
            v16bf fBr = __builtin_bit_cast(v16bf, br);
            v16bf fBi = __builtin_bit_cast(v16bf, bi);

            p1[t] = __builtin_amdgcn_wmma_f32_16x16x32_bf16(false, fAr, false, fBr, (short)0, p1[t], false, false);
            p2[t] = __builtin_amdgcn_wmma_f32_16x16x32_bf16(false, fAi, false, fBi, (short)0, p2[t], false, false);
            p3[t] = __builtin_amdgcn_wmma_f32_16x16x32_bf16(false, fAr, false, fBi, (short)0, p3[t], false, false);
            p4[t] = __builtin_amdgcn_wmma_f32_16x16x32_bf16(false, fAi, false, fBr, (short)0, p4[t], false, false);
        }
    }

    // Epilogue: combine chains, scale by gate score, atomic scatter-add.
    // C/D layout: lane l -> col N = l&15 ; VGPR v -> row M = v + 8*(l>>4)
#pragma unroll
    for (int t = 0; t < 4; ++t) {
        const int ncol = (waveId * 4 + t) * 16 + (lane & 15);
        v8f accR = p1[t] - p2[t];
        v8f accI = p3[t] + p4[t];
#pragma unroll
        for (int v = 0; v < 8; ++v) {
            const int M    = v + ((lane >> 4) ? 8 : 0);
            const int rsel = e * KSEL + mt * 16 + M;
            const int tok  = idx_all[rsel];
            const float sc = sc_all[rsel];
            float* dst = out + (size_t)tok * 1024 + (size_t)ncol * 2;
            atomicAdd(dst,     accR[v] * sc);
            atomicAdd(dst + 1, accI[v] * sc);
        }
    }
}

// ---------------- 6) divide by counts + ModReLU, emit counts ----------------
__global__ void finalize_kernel(const float* __restrict__ counts,
                                const float* __restrict__ bias,
                                float* __restrict__ out) {
    int i = blockIdx.x * 256 + threadIdx.x;           // i < B_T*DDIM
    int tok = i >> 9;
    int d   = i & 511;
    float re = out[(size_t)i * 2];
    float im = out[(size_t)i * 2 + 1];
    float c  = counts[tok];
    float denom = fmaxf(c, 1.0f);
    re /= denom; im /= denom;
    float mag = sqrtf(re * re + im * im);
    float act = fmaxf(mag + bias[d], 0.0f);
    float scl = act / fmaxf(mag, 1e-8f);
    out[(size_t)i * 2]     = re * scl;
    out[(size_t)i * 2 + 1] = im * scl;
    if (d == 0) out[CNT_OFF + tok] = c;
}

extern "C" void kernel_launch(void* const* d_in, const int* in_sizes, int n_in,
                              void* d_out, int out_size, void* d_ws, size_t ws_size,
                              hipStream_t stream) {
    const float* x    = (const float*)d_in[0];
    const float* gw   = (const float*)d_in[1];
    const float* w    = (const float*)d_in[2];
    const float* bias = (const float*)d_in[3];
    float* out = (float*)d_out;

    char* ws = (char*)d_ws;
    float*          scores = (float*)ws;                                    // 1 MB
    unsigned short* Xr     = (unsigned short*)(ws + 1048576);               // 16 MB
    unsigned short* Xi     = (unsigned short*)(ws + 1048576 + 16777216);    // 16 MB
    unsigned short* WrT    = (unsigned short*)(ws + 1048576 + 2*16777216);  // 8 MB
    unsigned short* WiT    = (unsigned short*)(ws + 1048576 + 2*16777216 + 8388608);
    float*          counts = (float*)(ws + 1048576 + 2*16777216 + 2*8388608);

    // zero accumulation region of out + counts (graph-capture safe)
    hipMemsetAsync(out, 0, (size_t)B_T * DDIM * 2 * sizeof(float), stream);
    hipMemsetAsync(counts, 0, (size_t)B_T * sizeof(float), stream);

    cvt_x_kernel<<<32768, 256, 0, stream>>>(x, Xr, Xi);          // 8.39M elems
    cvt_w_kernel<<<16384, 256, 0, stream>>>(w, WrT, WiT);        // 4.19M elems
    gate_kernel<<<64, 256, 0, stream>>>(x, gw, scores);          // 16384 tokens
    topk_kernel<<<NEXP, 1024, B_T * 8, stream>>>(scores, out, counts);
    moe_gemm_kernel<<<NEXP * 64, 256, 0, stream>>>(Xr, Xi, WrT, WiT, out);
    finalize_kernel<<<32768, 256, 0, stream>>>(counts, bias, out);
}
